// BertSwitchForSequenceClassification_9929964389212
// MI455X (gfx1250) — matrix operand order repaired and checked by reading, not compiled, not run on hardware
//
#include <hip/hip_runtime.h>
#include <hip/hip_bf16.h>
#include <math.h>

// ---------------- model constants ----------------
#define TT   4096      // B*S tokens
#define DD   768       // hidden
#define SS_  512       // seq len
#define BB   8         // batch
#define HH   12        // heads
#define HDIM 64        // head dim
#define EE   8         // experts
#define CC   640       // capacity = 1.25*T/E
#define FF_  3072      // ffn dim
#define LL   4         // layers
#define NL_  2         // num labels

typedef __attribute__((ext_vector_type(16))) _Float16 v16h;
typedef __attribute__((ext_vector_type(8)))  _Float16 v8h;
typedef __attribute__((ext_vector_type(4)))  _Float16 v4h;
typedef __attribute__((ext_vector_type(8)))  float    v8f;

// ---------------- generic batched WMMA GEMM ----------------
// C[M,N] = A[M,K] * B(K,N) (+bias) (ReLU opt). fp32 in/out, f16 WMMA compute.
// Block tile 128(M) x 64(N), K-step 32. 4 waves; each wave owns 32 rows:
// 2 A-fragments x 4 B-fragments = 8 WMMAs per K-step. All fragments preloaded
// into distinct registers before the WMMA burst so ds waits can be partial.
// LDS: As[buf][row][k], Bs[buf][col][k] (K-contig -> 2x ds_load_b128/fragment).
// Register double-buffer overlaps HBM loads with WMMA compute.
// Batch offset: off = (g/gdiv)*s1 + (g%gdiv)*s2 for A,B,C. bias off = g*biasStride.
__global__ __launch_bounds__(128)
void wmma_gemm_kernel(const float* __restrict__ A, const float* __restrict__ Bm,
                      const float* __restrict__ bias, float* __restrict__ Cm,
                      int M, int N, int K, int lda, int ldb, int ldc, int gdiv,
                      long long sA1, long long sA2, long long sB1, long long sB2,
                      long long sC1, long long sC2, long long biasStride,
                      int transB, int relu)
{
    __shared__ _Float16 As[2][128][40];  // [buf][row][k], 80B row stride (16B aligned chunks)
    __shared__ _Float16 Bs[2][64][40];   // [buf][col][k]

    const int g = blockIdx.z;
    const long long gq = g / gdiv, gr = g % gdiv;
    A  += gq * sA1 + gr * sA2;
    Bm += gq * sB1 + gr * sB2;
    Cm += gq * sC1 + gr * sC2;
    if (bias) bias += (long long)g * biasStride;

    const int n0   = blockIdx.x * 64;
    const int m0   = blockIdx.y * 128;
    const int tid  = threadIdx.x;
    const int wave = tid >> 5;
    const int lane = tid & 31;
    const int half = lane >> 4;
    const int l16  = lane & 15;

    const v8f vz = {0.f,0.f,0.f,0.f,0.f,0.f,0.f,0.f};
    v8f acc[2][4] = {{vz, vz, vz, vz}, {vz, vz, vz, vz}};

    float4 ra[8], rb[4];

    // A mapping: e = tid+i*128 (i<8) -> r = e>>3 (0..127), kq = e&7 (float4 along K)
    // B mapping: transB: c = e>>3, kq = e&7 (source K-contig -> float4)
    //            else  : c = e&63, kq = e>>6 (lane-adjacent cols -> coalesced b32)
#define LOAD_TILE(k0_)                                                          \
    _Pragma("unroll")                                                           \
    for (int i = 0; i < 8; ++i) {                                               \
        int e = tid + i * 128;                                                  \
        int r = e >> 3, kq = e & 7;                                             \
        ra[i] = *(const float4*)&A[(long long)(m0 + r) * lda + (k0_) + kq * 4]; \
    }                                                                           \
    _Pragma("unroll")                                                           \
    for (int i = 0; i < 4; ++i) {                                               \
        int e = tid + i * 128;                                                  \
        if (transB) {                                                           \
            int c = e >> 3, kq2 = e & 7;                                        \
            rb[i] = *(const float4*)&Bm[(long long)(n0 + c) * ldb + (k0_) + kq2 * 4]; \
        } else {                                                                \
            int c = e & 63, kq2 = e >> 6;                                       \
            float4 v;                                                           \
            v.x = Bm[(long long)((k0_) + kq2 * 4 + 0) * ldb + n0 + c];          \
            v.y = Bm[(long long)((k0_) + kq2 * 4 + 1) * ldb + n0 + c];          \
            v.z = Bm[(long long)((k0_) + kq2 * 4 + 2) * ldb + n0 + c];          \
            v.w = Bm[(long long)((k0_) + kq2 * 4 + 3) * ldb + n0 + c];          \
            rb[i] = v;                                                          \
        }                                                                       \
    }

#define STORE_TILE(p_)                                                          \
    _Pragma("unroll")                                                           \
    for (int i = 0; i < 8; ++i) {                                               \
        int e = tid + i * 128;                                                  \
        int r = e >> 3, kq = e & 7;                                             \
        v4h ha; ha[0] = (_Float16)ra[i].x; ha[1] = (_Float16)ra[i].y;           \
        ha[2] = (_Float16)ra[i].z; ha[3] = (_Float16)ra[i].w;                   \
        *(v4h*)&As[p_][r][kq * 4] = ha;                                         \
    }                                                                           \
    _Pragma("unroll")                                                           \
    for (int i = 0; i < 4; ++i) {                                               \
        int e = tid + i * 128;                                                  \
        int c, kq2;                                                             \
        if (transB) { c = e >> 3; kq2 = e & 7; } else { c = e & 63; kq2 = e >> 6; } \
        v4h hb; hb[0] = (_Float16)rb[i].x; hb[1] = (_Float16)rb[i].y;           \
        hb[2] = (_Float16)rb[i].z; hb[3] = (_Float16)rb[i].w;                   \
        *(v4h*)&Bs[p_][c][kq2 * 4] = hb;                                        \
    }

#define CAT16 0,1,2,3,4,5,6,7,8,9,10,11,12,13,14,15

    LOAD_TILE(0)
    STORE_TILE(0)
    int p = 0;
    for (int k0 = 0; k0 < K; k0 += 32) {
        __syncthreads();                      // tile p visible to all waves
        const bool more = (k0 + 32) < K;
        if (more) { LOAD_TILE(k0 + 32) }      // HBM loads overlap with WMMA below

        // preload ALL fragments (distinct regs) so WMMAs can issue back-to-back
        const int arow0 = wave * 32 + l16;
        v8h a0l = *(const v8h*)&As[p][arow0][half * 8];
        v8h a0h = *(const v8h*)&As[p][arow0][16 + half * 8];
        v8h a1l = *(const v8h*)&As[p][arow0 + 16][half * 8];
        v8h a1h = *(const v8h*)&As[p][arow0 + 16][16 + half * 8];
        v16h af0 = __builtin_shufflevector(a0l, a0h, CAT16);
        v16h af1 = __builtin_shufflevector(a1l, a1h, CAT16);
        v16h bf[4];
        #pragma unroll
        for (int nt = 0; nt < 4; ++nt) {
            const int col = nt * 16 + l16;
            v8h blo = *(const v8h*)&Bs[p][col][half * 8];
            v8h bhi = *(const v8h*)&Bs[p][col][16 + half * 8];
            bf[nt] = __builtin_shufflevector(blo, bhi, CAT16);
        }
        #pragma unroll
        for (int nt = 0; nt < 4; ++nt) {
            acc[0][nt] = __builtin_amdgcn_wmma_f32_16x16x32_f16(
                false, af0, false, bf[nt], (short)0, acc[0][nt], false, false);
            acc[1][nt] = __builtin_amdgcn_wmma_f32_16x16x32_f16(
                false, af1, false, bf[nt], (short)0, acc[1][nt], false, false);
        }
        __syncthreads();                      // everyone done reading p
        if (more) { STORE_TILE(p ^ 1) }
        p ^= 1;
    }
#undef LOAD_TILE
#undef STORE_TILE
#undef CAT16

    // epilogue: C/D layout — VGPR r: lanes0-15 M=r, lanes16-31 M=8+r
    #pragma unroll
    for (int sub = 0; sub < 2; ++sub) {
        #pragma unroll
        for (int nt = 0; nt < 4; ++nt) {
            const int col = n0 + nt * 16 + l16;
            const float bv = bias ? bias[col] : 0.f;
            #pragma unroll
            for (int r = 0; r < 8; ++r) {
                const int row = m0 + wave * 32 + sub * 16 + half * 8 + r;
                float v = acc[sub][nt][r] + bv;
                if (relu) v = fmaxf(v, 0.f);
                Cm[(long long)row * ldc + col] = v;
            }
        }
    }
    (void)M; (void)N;
}

// ---------------- block reduction helper (256 threads) ----------------
__device__ inline float block_reduce_sum256(float v, float* red) {
    int tid = threadIdx.x;
    red[tid] = v; __syncthreads();
    for (int s = 128; s > 0; s >>= 1) {
        if (tid < s) red[tid] += red[tid + s];
        __syncthreads();
    }
    float r = red[0]; __syncthreads();
    return r;
}

// ---------------- embeddings (float4, 192 threads) ----------------
__global__ __launch_bounds__(192)
void embed_kernel(const int* __restrict__ ids, const float* __restrict__ we,
                  const float* __restrict__ pe, const float* __restrict__ te,
                  float* __restrict__ x)
{
    int t = blockIdx.x;
    int tok = ids[t];
    int s = t & (SS_ - 1);
    int i = threadIdx.x;                       // 0..191 float4 lanes
    const float4 a = ((const float4*)(we + (long long)tok * DD))[i];
    const float4 b = ((const float4*)(pe + (long long)s * DD))[i];
    const float4 c = ((const float4*)te)[i];
    float4 o;
    o.x = a.x + b.x + c.x; o.y = a.y + b.y + c.y;
    o.z = a.z + b.z + c.z; o.w = a.w + b.w + c.w;
    ((float4*)(x + (long long)t * DD))[i] = o;
}

// ---------------- residual + LayerNorm (add==nullptr -> plain LN) ----------------
__global__ __launch_bounds__(256)
void add_ln_kernel(float* __restrict__ x, const float* __restrict__ add,
                   const float* __restrict__ g, const float* __restrict__ b)
{
    __shared__ float red[256];
    int t = blockIdx.x, tid = threadIdx.x;
    float4 v = {0.f, 0.f, 0.f, 0.f};
    if (tid < 192) {
        v = ((const float4*)(x + (long long)t * DD))[tid];
        if (add) {
            float4 a = ((const float4*)(add + (long long)t * DD))[tid];
            v.x += a.x; v.y += a.y; v.z += a.z; v.w += a.w;
        }
    }
    float s = v.x + v.y + v.z + v.w;
    float mu = block_reduce_sum256(s, red) * (1.0f / DD);
    float s2 = 0.f;
    if (tid < 192) {
        float dx = v.x - mu, dy = v.y - mu, dz = v.z - mu, dw = v.w - mu;
        s2 = dx * dx + dy * dy + dz * dz + dw * dw;
    }
    float var = block_reduce_sum256(s2, red) * (1.0f / DD);
    float rstd = rsqrtf(var + 1e-12f);
    if (tid < 192) {
        float4 gg = ((const float4*)g)[tid];
        float4 bb = ((const float4*)b)[tid];
        float4 o;
        o.x = (v.x - mu) * rstd * gg.x + bb.x;
        o.y = (v.y - mu) * rstd * gg.y + bb.y;
        o.z = (v.z - mu) * rstd * gg.z + bb.z;
        o.w = (v.w - mu) * rstd * gg.w + bb.w;
        ((float4*)(x + (long long)t * DD))[tid] = o;
    }
}

// ---------------- attention softmax (scale + mask, in place) ----------------
__global__ __launch_bounds__(256)
void softmax_kernel(float* __restrict__ scores, const int* __restrict__ mask, float scale)
{
    __shared__ float red[256];
    long long rowi = blockIdx.x;            // g*S + q,  g = b*H + h
    long long gg = rowi / SS_;
    int b = (int)(gg / HH);
    float* row = scores + rowi * SS_;
    int tid = threadIdx.x;
    float v0[2]; float lmax = -1e30f;
    #pragma unroll
    for (int i = 0; i < 2; ++i) {
        int c = tid + i * 256;
        float mb = (1.0f - (float)mask[b * SS_ + c]) * (-10000.0f);
        v0[i] = row[c] * scale + mb;
        lmax = fmaxf(lmax, v0[i]);
    }
    red[tid] = lmax; __syncthreads();
    for (int s = 128; s > 0; s >>= 1) {
        if (tid < s) red[tid] = fmaxf(red[tid], red[tid + s]);
        __syncthreads();
    }
    float m = red[0]; __syncthreads();
    float lsum = 0.f;
    #pragma unroll
    for (int i = 0; i < 2; ++i) { v0[i] = expf(v0[i] - m); lsum += v0[i]; }
    float tot = block_reduce_sum256(lsum, red);
    float inv = 1.0f / tot;
    #pragma unroll
    for (int i = 0; i < 2; ++i) row[tid + i * 256] = v0[i] * inv;
}

// ---------------- MoE router: softmax(x@Wr) -> top1 expert + gate ----------------
__global__ __launch_bounds__(64)
void router_kernel(const float* __restrict__ x, const float* __restrict__ Wr,
                   int* __restrict__ eidx, float* __restrict__ gate)
{
    __shared__ float red[64];
    __shared__ float tot8[EE];
    int t = blockIdx.x, tid = threadIdx.x;
    int e = tid & 7, part = tid >> 3;
    float s = 0.f;
    for (int i = part * 96; i < part * 96 + 96; ++i)
        s += x[(long long)t * DD + i] * Wr[i * EE + e];
    red[tid] = s; __syncthreads();
    if (tid < EE) {
        float tot = 0.f;
        for (int p = 0; p < 8; ++p) tot += red[tid + p * 8];
        tot8[tid] = tot;
    }
    __syncthreads();
    if (tid == 0) {
        float mx = tot8[0]; int am = 0;
        for (int j = 1; j < EE; ++j) if (tot8[j] > mx) { mx = tot8[j]; am = j; }
        float den = 0.f;
        for (int j = 0; j < EE; ++j) den += expf(tot8[j] - mx);
        eidx[t] = am;
        gate[t] = 1.0f / den;     // softmax prob at the argmax
    }
}

// ---------------- per-expert capacity scan (exact cumsum semantics) ----------------
__global__ void route_scan_kernel(const int* __restrict__ eidx, const float* __restrict__ gate,
                                  int* __restrict__ dest, int* __restrict__ keepi,
                                  float* __restrict__ scalef)
{
    int e = threadIdx.x;
    if (e >= EE) return;
    int cnt = 0;
    for (int t = 0; t < TT; ++t) {
        if (eidx[t] == e) {
            int pos = cnt++;
            int kp = (pos < CC) ? 1 : 0;
            dest[t]   = e * CC + (pos < CC ? pos : CC - 1);
            keepi[t]  = kp;
            scalef[t] = gate[t] * (float)kp;
        }
    }
}

// ---------------- dispatch / combine (float4, 192 threads) ----------------
__global__ __launch_bounds__(192)
void dispatch_kernel(const float* __restrict__ x, const int* __restrict__ dest,
                     const int* __restrict__ keepi, float* __restrict__ buf)
{
    int t = blockIdx.x;
    if (!keepi[t]) return;
    int i = threadIdx.x;
    ((float4*)(buf + (long long)dest[t] * DD))[i] =
        ((const float4*)(x + (long long)t * DD))[i];
}

__global__ __launch_bounds__(192)
void combine_kernel(const float* __restrict__ y, const int* __restrict__ dest,
                    const float* __restrict__ scalef, float* __restrict__ outb)
{
    int t = blockIdx.x;
    float s = scalef[t];
    int i = threadIdx.x;
    float4 v = ((const float4*)(y + (long long)dest[t] * DD))[i];
    v.x *= s; v.y *= s; v.z *= s; v.w *= s;
    ((float4*)(outb + (long long)t * DD))[i] = v;
}

// ---------------- pooled classifier + loss ----------------
__global__ __launch_bounds__(256)
void head_kernel(const float* __restrict__ x, const float* __restrict__ clsW,
                 const float* __restrict__ clsB, const int* __restrict__ labels,
                 float* __restrict__ out)
{
    __shared__ float pooled[BB * DD];
    __shared__ float lg[BB * NL_];
    int tid = threadIdx.x;
    for (int idx = tid; idx < BB * DD; idx += 256) {
        int b = idx / DD, i = idx % DD;
        float s = 0.f;
        for (int ss = 0; ss < SS_; ++ss)
            s += x[((long long)(b * SS_ + ss)) * DD + i];
        pooled[idx] = s * (1.0f / SS_);
    }
    __syncthreads();
    if (tid < BB * NL_) {
        int b = tid >> 1, c = tid & 1;
        float s = clsB[c];
        for (int i = 0; i < DD; ++i) s += pooled[b * DD + i] * clsW[i * NL_ + c];
        lg[tid] = s;
        out[1 + tid] = s;          // logits flat [B, NL]
    }
    __syncthreads();
    if (tid == 0) {
        float loss = 0.f;
        for (int b = 0; b < BB; ++b) {
            float l0 = lg[b * 2], l1 = lg[b * 2 + 1];
            float m = fmaxf(l0, l1);
            float lse = m + logf(expf(l0 - m) + expf(l1 - m));
            float lp = lg[b * 2 + labels[b]] - lse;
            loss -= lp * (1.0f / BB);
        }
        out[0] = loss;
    }
}

// ---------------- host side ----------------
static void gemm_launch(hipStream_t st, const float* A, const float* B, const float* bias,
                        float* C, int M, int N, int K, int lda, int ldb, int ldc,
                        int gdiv, long long sA1, long long sA2, long long sB1, long long sB2,
                        long long sC1, long long sC2, long long biasStride,
                        int transB, int relu, int batch)
{
    dim3 grid(N / 64, M / 128, batch);
    wmma_gemm_kernel<<<grid, 128, 0, st>>>(A, B, bias, C, M, N, K, lda, ldb, ldc,
                                           gdiv, sA1, sA2, sB1, sB2, sC1, sC2,
                                           biasStride, transB, relu);
}

extern "C" void kernel_launch(void* const* d_in, const int* in_sizes, int n_in,
                              void* d_out, int out_size, void* d_ws, size_t ws_size,
                              hipStream_t stream)
{
    const int*   input_ids = (const int*)d_in[0];
    const int*   attn_mask = (const int*)d_in[1];
    const int*   labels    = (const int*)d_in[2];
    const float* word_emb  = (const float*)d_in[3];
    const float* pos_emb   = (const float*)d_in[4];
    const float* type_emb  = (const float*)d_in[5];
    const float* emb_ln_g  = (const float*)d_in[6];
    const float* emb_ln_b  = (const float*)d_in[7];
    const float* Wq = (const float*)d_in[8];
    const float* bq = (const float*)d_in[9];
    const float* Wk = (const float*)d_in[10];
    const float* bk = (const float*)d_in[11];
    const float* Wv = (const float*)d_in[12];
    const float* bv = (const float*)d_in[13];
    const float* Wo = (const float*)d_in[14];
    const float* bo = (const float*)d_in[15];
    const float* ln1_g = (const float*)d_in[16];
    const float* ln1_b = (const float*)d_in[17];
    const float* Wr = (const float*)d_in[18];
    const float* W1 = (const float*)d_in[19];
    const float* b1 = (const float*)d_in[20];
    const float* W2 = (const float*)d_in[21];
    const float* b2 = (const float*)d_in[22];
    const float* ln2_g = (const float*)d_in[23];
    const float* ln2_b = (const float*)d_in[24];
    const float* cls_W = (const float*)d_in[25];
    const float* cls_b = (const float*)d_in[26];
    float* out = (float*)d_out;

    // carve fp32 workspace
    char*  ws  = (char*)d_ws;
    size_t off = 0;
    auto alloc = [&](size_t nfloats) -> void* {
        void* p = ws + off;
        off = (off + nfloats * 4 + 255) & ~(size_t)255;
        return p;
    };
    float* x    = (float*)alloc((size_t)TT * DD);
    float* qb   = (float*)alloc((size_t)TT * DD);
    float* kb   = (float*)alloc((size_t)TT * DD);
    float* vb   = (float*)alloc((size_t)TT * DD);
    float* sc   = (float*)alloc((size_t)BB * HH * SS_ * SS_);
    float* ctx  = (float*)alloc((size_t)TT * DD);
    float* tmp  = (float*)alloc((size_t)TT * DD);
    float* buf  = (float*)alloc((size_t)EE * CC * DD);
    float* hb   = (float*)alloc((size_t)EE * CC * FF_);
    float* yb   = (float*)alloc((size_t)EE * CC * DD);
    float* gate = (float*)alloc(TT);
    float* scal = (float*)alloc(TT);
    int*   eidx = (int*)alloc(TT);
    int*   dest = (int*)alloc(TT);
    int*   keep = (int*)alloc(TT);

    const long long sBH = (long long)SS_ * DD;   // per-batch row offset in [T,768]
    const long long sSS = (long long)SS_ * SS_;  // per-head score tile

    // embeddings + LN
    embed_kernel<<<TT, 192, 0, stream>>>(input_ids, word_emb, pos_emb, type_emb, x);
    add_ln_kernel<<<TT, 256, 0, stream>>>(x, nullptr, emb_ln_g, emb_ln_b);

    for (int l = 0; l < LL; ++l) {
        const float* Wq_l = Wq + (size_t)l * DD * DD;
        const float* Wk_l = Wk + (size_t)l * DD * DD;
        const float* Wv_l = Wv + (size_t)l * DD * DD;
        const float* Wo_l = Wo + (size_t)l * DD * DD;
        const float* bq_l = bq + (size_t)l * DD;
        const float* bk_l = bk + (size_t)l * DD;
        const float* bv_l = bv + (size_t)l * DD;
        const float* bo_l = bo + (size_t)l * DD;

        // Q,K,V projections: [4096,768] = x @ W + b
        gemm_launch(stream, x, Wq_l, bq_l, qb, TT, DD, DD, DD, DD, DD,
                    1, 0,0, 0,0, 0,0, 0, 0, 0, 1);
        gemm_launch(stream, x, Wk_l, bk_l, kb, TT, DD, DD, DD, DD, DD,
                    1, 0,0, 0,0, 0,0, 0, 0, 0, 1);
        gemm_launch(stream, x, Wv_l, bv_l, vb, TT, DD, DD, DD, DD, DD,
                    1, 0,0, 0,0, 0,0, 0, 0, 0, 1);

        // scores[b,h] = Q_bh [512,64] @ K_bh^T [64,512] ; batch g=b*H+h
        gemm_launch(stream, qb, kb, nullptr, sc, SS_, SS_, HDIM, DD, DD, SS_,
                    HH, sBH, HDIM, sBH, HDIM, (long long)HH * sSS, sSS,
                    0, 1, 0, BB * HH);

        // softmax with scale + mask (in place)
        softmax_kernel<<<BB * HH * SS_, 256, 0, stream>>>(sc, attn_mask, 0.125f);

        // ctx[b,h] = att [512,512] @ V_bh [512,64] -> strided into ctx[T,768]
        gemm_launch(stream, sc, vb, nullptr, ctx, SS_, HDIM, SS_, SS_, DD, DD,
                    HH, (long long)HH * sSS, sSS, sBH, HDIM, sBH, HDIM,
                    0, 0, 0, BB * HH);

        // output projection + residual LN
        gemm_launch(stream, ctx, Wo_l, bo_l, tmp, TT, DD, DD, DD, DD, DD,
                    1, 0,0, 0,0, 0,0, 0, 0, 0, 1);
        add_ln_kernel<<<TT, 256, 0, stream>>>(x, tmp, ln1_g + (size_t)l * DD, ln1_b + (size_t)l * DD);

        // --- Switch MoE ---
        router_kernel<<<TT, 64, 0, stream>>>(x, Wr + (size_t)l * DD * EE, eidx, gate);
        route_scan_kernel<<<1, 32, 0, stream>>>(eidx, gate, dest, keep, scal);
        hipMemsetAsync(buf, 0, (size_t)EE * CC * DD * 4, stream);
        dispatch_kernel<<<TT, 192, 0, stream>>>(x, dest, keep, buf);

        // h = relu(buf @ W1 + b1) per expert: [640,3072]
        gemm_launch(stream, buf, W1 + (size_t)l * EE * DD * FF_, b1 + (size_t)l * EE * FF_, hb,
                    CC, FF_, DD, DD, FF_, FF_,
                    1, (long long)CC * DD, 0, (long long)DD * FF_, 0,
                    (long long)CC * FF_, 0, FF_, 0, 1, EE);
        // y = h @ W2 + b2 per expert: [640,768]
        gemm_launch(stream, hb, W2 + (size_t)l * EE * FF_ * DD, b2 + (size_t)l * EE * DD, yb,
                    CC, DD, FF_, FF_, DD, DD,
                    1, (long long)CC * FF_, 0, (long long)FF_ * DD, 0,
                    (long long)CC * DD, 0, DD, 0, 0, EE);

        combine_kernel<<<TT, 192, 0, stream>>>(yb, dest, scal, tmp);
        add_ln_kernel<<<TT, 256, 0, stream>>>(x, tmp, ln2_g + (size_t)l * DD, ln2_b + (size_t)l * DD);
    }

    // mean-pool -> classifier -> log-softmax loss
    head_kernel<<<1, 256, 0, stream>>>(x, cls_W, cls_b, labels, out);

    (void)in_sizes; (void)n_in; (void)out_size; (void)ws_size;
}
